// MultiHeadAttention_73675868995782
// MI455X (gfx1250) — matrix operand order repaired and checked by reading, not compiled, not run on hardware
//
#include <hip/hip_runtime.h>

// ---------------------------------------------------------------------------
// MHA forward for MI455X (gfx1250, wave32, WMMA bf16 16x16x32, f32 accum)
// Data movement: GLOBAL_LOAD_ASYNC_TO_LDS (GEMM staging) + TDM tensor_load
// (attention K tiles, with TDM LDS padding to a 136-elem row stride).
// ---------------------------------------------------------------------------

typedef __bf16 bf16_t;
typedef __attribute__((ext_vector_type(8)))  __bf16 v8bf;
typedef __attribute__((ext_vector_type(16))) __bf16 v16bf;
typedef __attribute__((ext_vector_type(8)))  float  v8f;

#define HIDDEN   2048
#define HEADS    16
#define HEAD_DIM 128
#define BATCH    2
#define SEQ      2048
#define MROWS    (BATCH * SEQ)

#if __has_builtin(__builtin_amdgcn_global_load_async_to_lds_b128)
#define HAS_ASYNC_LDS 1
#else
#define HAS_ASYNC_LDS 0
#endif

#if __has_builtin(__builtin_amdgcn_tensor_load_to_lds)
#define HAS_TDM 1
#else
#define HAS_TDM 0
#endif

// exact parameter types for the async-LDS builtin (from hipcc diagnostic):
// param0 = global (AS1) pointer to 16B int vector, param1 = LDS (AS3) pointer
typedef int v4i_vs __attribute__((vector_size(4 * sizeof(int))));
typedef __attribute__((address_space(1))) v4i_vs* gptr_b128;
typedef __attribute__((address_space(3))) v4i_vs* lptr_b128;
typedef __attribute__((address_space(3))) bf16_t* as3_bf16p;

// TDM descriptor group types (this toolchain: 6-arg tensor_load_to_lds,
// i.e. (uint32x4, int32x8, int32x4, int32x4, int32x8, i32 cpol))
typedef unsigned int v4u_vs __attribute__((vector_size(16)));
typedef int          v8i_vs __attribute__((vector_size(32)));

// Async 16B copy global -> LDS (GLOBAL_LOAD_ASYNC_TO_LDS_B128, ASYNCcnt),
// falling back to load+store through VGPRs when the builtin is unavailable.
static __device__ __forceinline__ void copy16_g2l(bf16_t* lds_dst, const bf16_t* gsrc) {
#if HAS_ASYNC_LDS
    __builtin_amdgcn_global_load_async_to_lds_b128(
        (gptr_b128)gsrc, (lptr_b128)lds_dst, 0, 0);
#else
    *(v8bf*)lds_dst = *(const v8bf*)gsrc;
#endif
}

static __device__ __forceinline__ void async_wait0() {
#if HAS_ASYNC_LDS
#if __has_builtin(__builtin_amdgcn_s_wait_asynccnt)
    __builtin_amdgcn_s_wait_asynccnt(0);
#else
    asm volatile("s_wait_asynccnt 0" ::: "memory");
#endif
#endif
}

static __device__ __forceinline__ void tensor_wait0() {
#if __has_builtin(__builtin_amdgcn_s_wait_tensorcnt)
    __builtin_amdgcn_s_wait_tensorcnt(0);
#else
    asm volatile("s_wait_tensorcnt 0" ::: "memory");
#endif
}

#if HAS_TDM
// TDM load of a 2-D bf16 tile [rows x 128] from a row-major tensor with row
// stride 128 elems into LDS, padded by TDM to a 136-elem LDS row stride
// (pad_interval = 64 DWORDs -> code 5, pad_amount = 4 DWORDs -> code 3).
static __device__ __forceinline__ void tdm_load_tile_128x(unsigned lds_byte_off,
                                                          const bf16_t* gsrc,
                                                          int rows) {
    unsigned long long ga = (unsigned long long)gsrc;
    v4u_vs g0;
    g0[0] = 1u;                                   // count=1 (valid D#)
    g0[1] = lds_byte_off;                         // lds_addr
    g0[2] = (unsigned)(ga & 0xFFFFFFFFull);       // global_addr[31:0]
    g0[3] = (unsigned)((ga >> 32) & 0x1FFFFFFull) // global_addr[56:32]
            | (2u << 30);                         // type=2 ("image")
    v8i_vs g1;
    g1[0] = (int)(0x00010000u                     // data_size=1 -> 2 bytes
            | (1u << 20)                          // pad_enable
            | (5u << 22)                          // pad_interval: 64 DWORDs
            | (3u << 25));                        // pad_amount:   4 DWORDs
    g1[1] = (int)(128u << 16);                    // tensor_dim0 = 128 (bits 79:48)
    g1[2] = (int)(2048u << 16);                   // tensor_dim1 = 2048 (bits 111:80)
    g1[3] = (int)(128u << 16);                    // tile_dim0 = 128 (bits 127:112)
    g1[4] = rows;                                 // tile_dim1 (bits 143:128)
    g1[5] = 128;                                  // tensor_dim0_stride = 128 elems
    g1[6] = 0;
    g1[7] = 0;
    v4i_vs z4 = {0, 0, 0, 0};                     // groups 2/3 unused (2-D tile)
    v8i_vs z8 = {0, 0, 0, 0, 0, 0, 0, 0};         // trailing group: zero-filled
    __builtin_amdgcn_tensor_load_to_lds(g0, g1, z4, z4, z8, 0);
}
#endif

// round-to-nearest-even f32 -> bf16 (bit twiddle; no __bf16 arithmetic)
static __device__ __forceinline__ bf16_t to_bf16(float f) {
    unsigned u = __float_as_uint(f);
    unsigned r = u + 0x7FFFu + ((u >> 16) & 1u);
    unsigned short h = (unsigned short)(r >> 16);
    return __builtin_bit_cast(__bf16, h);
}

static __device__ __forceinline__ v16bf cat8(v8bf lo, v8bf hi) {
    v16bf r;
#pragma unroll
    for (int j = 0; j < 8; ++j) { r[j] = lo[j]; r[j + 8] = hi[j]; }
    return r;
}

static __device__ __forceinline__ v8f wmma_bf16(v16bf a, v16bf b, v8f c) {
    // D = A(16x32) * B(32x16) + C, f32 accum
    return __builtin_amdgcn_wmma_f32_16x16x32_bf16(
        /*neg_a=*/false, a, /*neg_b=*/false, b,
        /*c_mod=*/(short)0, c, /*reuse_a=*/false, /*reuse_b=*/false);
}

// ---------------------------------------------------------------------------
// fp32 -> bf16 conversion (grid-stride)
// ---------------------------------------------------------------------------
__global__ __launch_bounds__(256)
void cvt_f32_bf16(const float* __restrict__ s, bf16_t* __restrict__ d, int n) {
    int i = blockIdx.x * blockDim.x + threadIdx.x;
    int stride = gridDim.x * blockDim.x;
    for (; i < n; i += stride) d[i] = to_bf16(s[i]);
}

// ---------------------------------------------------------------------------
// GEMM: Y[M,N] = A[M,K] * W[N,K]^T + bias[N]
//   MODE 0: write fp32 to outF[m*N+n]          (final projection)
//   MODE 1: write bf16 to outH in [B,H,S,D]    (Q/K/V projections)
// block tile 128(M) x 64(N), 256 threads = 8 waves, each wave 32x32
// A/B staged into LDS via GLOBAL_LOAD_ASYNC_TO_LDS_B128 when available.
// ---------------------------------------------------------------------------
template <int MODE>
__global__ __launch_bounds__(256)
void gemm_bf16(const bf16_t* __restrict__ A, const bf16_t* __restrict__ Bw,
               const float* __restrict__ bias, float* __restrict__ outF,
               bf16_t* __restrict__ outH)
{
    constexpr int K   = HIDDEN;
    constexpr int N   = HIDDEN;
    constexpr int LDS_STRIDE = 40;   // 32 + pad (bf16 elems), keeps 16B align

    __shared__ bf16_t As[128 * LDS_STRIDE];
    __shared__ bf16_t Bs[64  * LDS_STRIDE];

    const int tid  = threadIdx.x;
    const int lane = tid & 31;
    const int wv   = tid >> 5;
    const int l16  = lane & 15;
    const int half = lane >> 4;
    const int wm   = wv & 3;        // 4 wave-tiles along M
    const int wn   = wv >> 2;       // 2 wave-tiles along N

    const int m0 = blockIdx.y * 128;
    const int n0 = blockIdx.x * 64;

    v8f c[2][2] = {};

    const int ar = tid >> 1, ac = (tid & 1) * 16;   // A stage: 128x32, 16/thr
    const int br = tid >> 2, bc = (tid & 3) * 8;    // B stage:  64x32,  8/thr

    for (int k0 = 0; k0 < K; k0 += 32) {
        __syncthreads();
        {
            const bf16_t* gp = A + (size_t)(m0 + ar) * K + k0 + ac;
            copy16_g2l(&As[ar * LDS_STRIDE + ac],     gp);
            copy16_g2l(&As[ar * LDS_STRIDE + ac + 8], gp + 8);
        }
        {
            const bf16_t* gp = Bw + (size_t)(n0 + br) * K + k0 + bc;
            copy16_g2l(&Bs[br * LDS_STRIDE + bc], gp);
        }
        if (k0 + 32 < K) {
            __builtin_prefetch(A  + (size_t)(m0 + ar) * K + k0 + 32 + ac, 0, 1);
            __builtin_prefetch(Bw + (size_t)(n0 + br) * K + k0 + 32 + bc, 0, 1);
        }
        async_wait0();
        __syncthreads();

        v16bf a[2], b[2];
#pragma unroll
        for (int i = 0; i < 2; ++i) {
            // A-frag: row = l16 (within 16-tile), K slots {kb..kb+7, kb+16..kb+23}
            const bf16_t* p = &As[(wm * 32 + i * 16 + l16) * LDS_STRIDE + half * 8];
            a[i] = cat8(*(const v8bf*)p, *(const v8bf*)(p + 16));
        }
#pragma unroll
        for (int j = 0; j < 2; ++j) {
            // B-frag: col n = l16, K slots {kb..kb+7, kb+16..kb+23}, kb = half*8
            const bf16_t* p = &Bs[(wn * 32 + j * 16 + l16) * LDS_STRIDE + half * 8];
            b[j] = cat8(*(const v8bf*)p, *(const v8bf*)(p + 16));
        }
#pragma unroll
        for (int i = 0; i < 2; ++i)
#pragma unroll
            for (int j = 0; j < 2; ++j)
                c[i][j] = wmma_bf16(a[i], b[j], c[i][j]);
    }

    // epilogue: C layout -> lane holds (m = r + 8*half, n = l16) per v8f slot r
#pragma unroll
    for (int i = 0; i < 2; ++i) {
#pragma unroll
        for (int j = 0; j < 2; ++j) {
#pragma unroll
            for (int r = 0; r < 8; ++r) {
                int m = m0 + wm * 32 + i * 16 + r + 8 * half;
                int n = n0 + wn * 32 + j * 16 + l16;
                float v = c[i][j][r] + bias[n];
                if (MODE == 0) {
                    outF[(size_t)m * N + n] = v;
                } else {
                    int bb = m >> 11;            // m = b*SEQ + s
                    int s  = m & (SEQ - 1);
                    int h  = n >> 7;             // n = h*HEAD_DIM + d
                    int d  = n & (HEAD_DIM - 1);
                    outH[(((size_t)bb * HEADS + h) * SEQ + s) * HEAD_DIM + d] =
                        to_bf16(v);
                }
            }
        }
    }
}

// ---------------------------------------------------------------------------
// Flash-style attention: softmax(Q K^T / sqrt(d)) V  per (b,h)
// 128 threads = 4 waves; each wave owns a 16-row Q block.
// K chunk (32x128) staged by the Tensor Data Mover (wave 0 issues one
// tensor_load_to_lds, TDM pads rows to 136 elems); V transposed manually.
// Writes ctx as bf16 in [B, S, HIDDEN].
// ---------------------------------------------------------------------------
__global__ __launch_bounds__(128)
void attention_fwd(const bf16_t* __restrict__ Q, const bf16_t* __restrict__ Kb,
                   const bf16_t* __restrict__ Vb, bf16_t* __restrict__ Ctx)
{
    constexpr int D   = HEAD_DIM;
    constexpr int LDK = 136;   // Ks row stride (128 + 8 pad; matches TDM padding)
    constexpr int LDV = 40;    // Vt row stride (32 keys + pad)
    constexpr int LDP = 40;    // P buffer row stride

    __shared__ bf16_t Ks[32 * LDK];      // [key][d]
    __shared__ bf16_t Vt[D * LDV];       // [d][key]  (transposed V chunk)
    __shared__ bf16_t Ps[4][16 * LDP];   // per-wave P tile [qrow][key]

    const int tid  = threadIdx.x;
    const int lane = tid & 31;
    const int wv   = tid >> 5;
    const int l16  = lane & 15;
    const int half = lane >> 4;

    const int bh = blockIdx.y;                 // b*HEADS + h
    const int q0 = blockIdx.x * 64 + wv * 16;  // this wave's first q row

    const bf16_t* qbase = Q  + ((size_t)bh * SEQ) * D;
    const bf16_t* kbase = Kb + ((size_t)bh * SEQ) * D;
    const bf16_t* vbase = Vb + ((size_t)bh * SEQ) * D;

    // resident Q fragments: 16 x 128 as 4 A-fragments (16x32 each)
    v16bf qf[4];
#pragma unroll
    for (int i = 0; i < 4; ++i) {
        const bf16_t* p = qbase + (size_t)(q0 + l16) * D + i * 32 + half * 8;
        qf[i] = cat8(*(const v8bf*)p, *(const v8bf*)(p + 16));
    }

    v8f acc[8] = {};
    float mrow[8], lrow[8];
#pragma unroll
    for (int r = 0; r < 8; ++r) { mrow[r] = -__builtin_inff(); lrow[r] = 0.f; }

    const float rscale = 0.08838834764831845f;   // 1/sqrt(128)

    const int key = tid >> 2;          // staging: 0..31
    const int dg  = (tid & 3) * 32;    // staging: 0,32,64,96

    for (int ck = 0; ck < SEQ / 32; ++ck) {
        const int kglob = ck * 32;
        __syncthreads();
        {
            const bf16_t* vp = vbase + (size_t)(kglob + key) * D + dg;
#pragma unroll
            for (int g = 0; g < 4; ++g) {
                // V chunk: transpose while staging ([d][key])
                v8bf vvv = *(const v8bf*)(vp + g * 8);
#pragma unroll
                for (int e = 0; e < 8; ++e)
                    Vt[(dg + g * 8 + e) * LDV + key] = vvv[e];
            }
            if (kglob + 32 < SEQ) __builtin_prefetch(vp + 32 * D, 0, 1);
        }
#if HAS_TDM
        // K chunk: one TDM descriptor load issued by wave 0 (per-wave DMA)
        if (wv == 0) {
            unsigned lds_off =
                (unsigned)(unsigned long long)(as3_bf16p)&Ks[0];
            tdm_load_tile_128x(lds_off, kbase + (size_t)kglob * D, 32);
            tensor_wait0();
        }
#else
        {
            const bf16_t* kp = kbase + (size_t)(kglob + key) * D + dg;
#pragma unroll
            for (int g = 0; g < 4; ++g)
                copy16_g2l(&Ks[key * LDK + dg + g * 8], kp + g * 8);
            async_wait0();
        }
#endif
        __syncthreads();

        // preload all K B-fragments, then run the WMMA chain (overlaps LDS
        // latency with matrix ops instead of load->wait->wmma lockstep)
        v16bf kbf[2][4];
#pragma unroll
        for (int i = 0; i < 4; ++i) {
            const bf16_t* p0 = &Ks[(l16)      * LDK + i * 32 + half * 8];
            const bf16_t* p1 = &Ks[(16 + l16) * LDK + i * 32 + half * 8];
            kbf[0][i] = cat8(*(const v8bf*)p0, *(const v8bf*)(p0 + 16));
            kbf[1][i] = cat8(*(const v8bf*)p1, *(const v8bf*)(p1 + 16));
        }
        v8f s0 = {}, s1 = {};
#pragma unroll
        for (int i = 0; i < 4; ++i) s0 = wmma_bf16(qf[i], kbf[0][i], s0);
#pragma unroll
        for (int i = 0; i < 4; ++i) s1 = wmma_bf16(qf[i], kbf[1][i], s1);

#pragma unroll
        for (int r = 0; r < 8; ++r) { s0[r] = s0[r] * rscale; s1[r] = s1[r] * rscale; }

        // online softmax: reduce across the 16 lanes of each half-wave
        float cm[8], rs[8], scl[8];
#pragma unroll
        for (int r = 0; r < 8; ++r) cm[r] = fmaxf(s0[r], s1[r]);
#pragma unroll
        for (int msk = 1; msk < 16; msk <<= 1)
#pragma unroll
            for (int r = 0; r < 8; ++r)
                cm[r] = fmaxf(cm[r], __shfl_xor(cm[r], msk, 16));
#pragma unroll
        for (int r = 0; r < 8; ++r) {
            float nm = fmaxf(mrow[r], cm[r]);
            scl[r]  = __expf(mrow[r] - nm);
            mrow[r] = nm;
            s0[r] = __expf(s0[r] - nm);
            s1[r] = __expf(s1[r] - nm);
            rs[r] = s0[r] + s1[r];
        }
#pragma unroll
        for (int msk = 1; msk < 16; msk <<= 1)
#pragma unroll
            for (int r = 0; r < 8; ++r)
                rs[r] += __shfl_xor(rs[r], msk, 16);
#pragma unroll
        for (int r = 0; r < 8; ++r) lrow[r] = lrow[r] * scl[r] + rs[r];

        // C-layout -> A-layout redistribution of P through per-wave LDS
#pragma unroll
        for (int r = 0; r < 8; ++r) {
            int mm = r + 8 * half;
            Ps[wv][mm * LDP + l16]      = to_bf16(s0[r]);
            Ps[wv][mm * LDP + 16 + l16] = to_bf16(s1[r]);
        }
        const bf16_t* pp = &Ps[wv][l16 * LDP + half * 8];
        v16bf pa = cat8(*(const v8bf*)pp, *(const v8bf*)(pp + 16));

        // rescale ctx accumulators, then ctx += P @ V (V-frags preloaded 4-wide)
#pragma unroll
        for (int jj = 0; jj < 2; ++jj) {
            v16bf vbf[4];
#pragma unroll
            for (int j = 0; j < 4; ++j) {
                const bf16_t* vp2 = &Vt[((jj * 4 + j) * 16 + l16) * LDV + half * 8];
                vbf[j] = cat8(*(const v8bf*)vp2, *(const v8bf*)(vp2 + 16));
            }
#pragma unroll
            for (int j = 0; j < 4; ++j) {
                int t = jj * 4 + j;
#pragma unroll
                for (int r = 0; r < 8; ++r) acc[t][r] = acc[t][r] * scl[r];
                acc[t] = wmma_bf16(pa, vbf[j], acc[t]);
            }
        }
    }

    // normalize and write ctx bf16 in [B, S, HIDDEN]
    const int bb = bh >> 4;
    const int h  = bh & (HEADS - 1);
#pragma unroll
    for (int j = 0; j < 8; ++j) {
#pragma unroll
        for (int r = 0; r < 8; ++r) {
            int srow = q0 + r + 8 * half;
            int col  = h * D + j * 16 + l16;
            float v = acc[j][r] / lrow[r];
            Ctx[((size_t)bb * SEQ + srow) * HIDDEN + col] = to_bf16(v);
        }
    }
}

// ---------------------------------------------------------------------------
// launcher
// ---------------------------------------------------------------------------
extern "C" void kernel_launch(void* const* d_in, const int* in_sizes, int n_in,
                              void* d_out, int out_size, void* d_ws, size_t ws_size,
                              hipStream_t stream)
{
    (void)in_sizes; (void)n_in; (void)out_size; (void)ws_size;

    const float* x  = (const float*)d_in[0];
    const float* wq = (const float*)d_in[1];
    const float* bq = (const float*)d_in[2];
    const float* wk = (const float*)d_in[3];
    const float* bk = (const float*)d_in[4];
    const float* wvw= (const float*)d_in[5];
    const float* bv = (const float*)d_in[6];
    const float* wo = (const float*)d_in[7];
    const float* bo = (const float*)d_in[8];
    float* out = (float*)d_out;

    const size_t NX = (size_t)BATCH * SEQ * HIDDEN;   // 8,388,608
    const size_t NW = (size_t)HIDDEN * HIDDEN;        // 4,194,304

    char* p = (char*)d_ws;
    bf16_t* xb   = (bf16_t*)p; p += NX * 2;
    bf16_t* wqb  = (bf16_t*)p; p += NW * 2;
    bf16_t* wkb  = (bf16_t*)p; p += NW * 2;
    bf16_t* wvb  = (bf16_t*)p; p += NW * 2;
    bf16_t* wob  = (bf16_t*)p; p += NW * 2;
    bf16_t* qb   = (bf16_t*)p; p += NX * 2;
    bf16_t* kbuf = (bf16_t*)p; p += NX * 2;
    bf16_t* vbuf = (bf16_t*)p; p += NX * 2;
    bf16_t* ctxb = (bf16_t*)p; p += NX * 2;

    cvt_f32_bf16<<<2048, 256, 0, stream>>>(x,   xb,  (int)NX);
    cvt_f32_bf16<<<2048, 256, 0, stream>>>(wq,  wqb, (int)NW);
    cvt_f32_bf16<<<2048, 256, 0, stream>>>(wk,  wkb, (int)NW);
    cvt_f32_bf16<<<2048, 256, 0, stream>>>(wvw, wvb, (int)NW);
    cvt_f32_bf16<<<2048, 256, 0, stream>>>(wo,  wob, (int)NW);

    dim3 gB(256), gG(HIDDEN / 64, MROWS / 128);
    gemm_bf16<1><<<gG, gB, 0, stream>>>(xb, wqb, bq, nullptr, qb);
    gemm_bf16<1><<<gG, gB, 0, stream>>>(xb, wkb, bk, nullptr, kbuf);
    gemm_bf16<1><<<gG, gB, 0, stream>>>(xb, wvb, bv, nullptr, vbuf);

    dim3 aB(128), aG(SEQ / 64, BATCH * HEADS);
    attention_fwd<<<aG, aB, 0, stream>>>(qb, kbuf, vbuf, ctxb);

    gemm_bf16<0><<<gG, gB, 0, stream>>>(ctxb, wob, bo, out, nullptr);
}
